// ToroidalMLP_46084999086218
// MI455X (gfx1250) — matrix operand order repaired and checked by reading, not compiled, not run on hardware
//
#include <hip/hip_runtime.h>

// ---------------------------------------------------------------------------
// ToroidalMLP fused kernel for gfx1250 (MI455X).
//   B=8192, IN=H=512, OUT=256, K_hist=3, steps=30 (runtime), ALPHA=0.5
// One workgroup (256 threads = 8 wave32) owns a 32-row batch tile and runs the
// entire 30-step recurrence out of an LDS ring of four bf16 state buffers.
// All GEMMs use v_wmma_f32_16x16x32_bf16 with fp32 accumulation.
// ---------------------------------------------------------------------------

typedef __attribute__((ext_vector_type(8)))  float  v8f;
typedef __attribute__((ext_vector_type(8)))  __bf16 v8bf;
typedef __attribute__((ext_vector_type(16))) __bf16 v16bf;

#define IN_DIM   512
#define H_DIM    512
#define OUT_DIM  256
#define MTILE    32
#define LDS_STRIDE 520                      // 512 + 8 halves pad: conflict-free ds_load_b128
#define SLOT_USHORTS (MTILE * LDS_STRIDE)   // one ring slot (bf16 elements)
#define RING_BYTES (4 * SLOT_USHORTS * 2)   // 133,120 B dynamic LDS (<= 320 KB/WGP)

static __device__ __forceinline__ unsigned short f2bf(float f) {
  unsigned u = __builtin_bit_cast(unsigned, f);
  u += 0x7FFFu + ((u >> 16) & 1u);          // round-to-nearest-even
  return (unsigned short)(u >> 16);
}
static __device__ __forceinline__ float bf2f(unsigned short h) {
  return __builtin_bit_cast(float, (unsigned)h << 16);
}
static __device__ __forceinline__ float fast_tanh(float x) {
  // tanh(x) = 1 - 2/(exp(2x)+1); exp overflow/underflow saturates to +/-1 correctly
  float e = __expf(2.0f * x);
  return 1.0f - 2.0f / (e + 1.0f);
}

// Build the 16x32 bf16 A-fragment for lane (l16, lh) from a row-major bf16
// matrix with row stride `stride` halves (contiguous-k order, not the ISA table order):
//   elems 0..15 = A[row][k0 + lh*16 + j] (contiguous k, matches B fragments)
static __device__ __forceinline__ v16bf load_a_frag(const unsigned short* base) {
  v8bf lo = *(const v8bf*)base;
  v8bf hi = *(const v8bf*)(base + 8);
  return __builtin_shufflevector(lo, hi, 0, 1, 2, 3, 4, 5, 6, 7,
                                         8, 9, 10, 11, 12, 13, 14, 15);
}

__global__ void weights_to_bf16(const float* __restrict__ src,
                                unsigned short* __restrict__ dst, int n) {
  int i = blockIdx.x * blockDim.x + threadIdx.x;
  if (i < n) dst[i] = f2bf(src[i]);
}

__global__ __launch_bounds__(256, 1)
void toroidal_mlp_kernel(const float* __restrict__ xg,
                         const unsigned short* __restrict__ Wx_bf,
                         const float* __restrict__ Wx_b,
                         const unsigned short* __restrict__ Wh_bf,
                         const float* __restrict__ Wh_b,
                         const unsigned short* __restrict__ Hd_bf,
                         const float* __restrict__ Hd_b,
                         const float* __restrict__ gamma_p,
                         const float* __restrict__ alphas_p,
                         const int* __restrict__ steps_p,
                         float* __restrict__ out) {
  extern __shared__ unsigned short ring[];  // 4 slots of [MTILE][LDS_STRIDE] bf16

  const int tid  = threadIdx.x;
  const int wave = tid >> 5;
  const int lane = tid & 31;
  const int l16  = lane & 15;
  const int lh   = lane >> 4;               // K-half selector (wave32 WMMA layout)
  const int wg   = blockIdx.x;              // batch rows [wg*32, wg*32+32)

  const int mstrip = wave & 1;              // rows 0-15 / 16-31 of the tile
  const int nbase  = (wave >> 1) * 128;     // 8 waves => full 32 x 512 per pass
  const int arow   = mstrip * 16 + l16;     // this lane's A-matrix row

  // ---- zero history slots 1..3 ----------------------------------------
  unsigned int* ru32 = (unsigned int*)ring;
  for (int i = tid; i < 3 * SLOT_USHORTS / 2; i += 256)
    ru32[SLOT_USHORTS / 2 + i] = 0u;

  // ---- x tile -> bf16 into ring slot 0 ---------------------------------
  const float* xtile = xg + (size_t)wg * MTILE * IN_DIM;
  for (int i = tid; i < MTILE * IN_DIM; i += 256) {
    int r = i >> 9, c = i & 511;
    ring[r * LDS_STRIDE + c] = f2bf(xtile[i]);
  }
  __syncthreads();

  // ---- phase 0: xproj = x @ Wx^T + Wx_b (stays in registers) -----------
  v8f acc[8];
#pragma unroll
  for (int nt = 0; nt < 8; ++nt) acc[nt] = v8f{};

  for (int kc = 0; kc < 16; ++kc) {
    v16bf a = load_a_frag(&ring[arow * LDS_STRIDE + kc * 32 + lh * 16]);
#pragma unroll
    for (int nt = 0; nt < 8; ++nt) {
      // B-fragment: B[k][n] = Wx[n][k]; lane holds 16 contiguous k of row n
      const __bf16* bp = (const __bf16*)Wx_bf +
                         (nbase + nt * 16 + l16) * IN_DIM + kc * 32 + lh * 16;
      acc[nt] = __builtin_amdgcn_wmma_f32_16x16x32_bf16(
          false, a, false, *(const v16bf*)bp, (short)0, acc[nt], false, false);
    }
  }

  float xp[8][8];
#pragma unroll
  for (int nt = 0; nt < 8; ++nt) {
    float bias = Wx_b[nbase + nt * 16 + l16];
#pragma unroll
    for (int r = 0; r < 8; ++r) xp[nt][r] = acc[nt][r] + bias;
  }
  __syncthreads();                          // everyone done reading slot 0

  // ---- zero slot 0 (s^(0) = 0), preload per-lane loop constants --------
  for (int i = tid; i < SLOT_USHORTS / 2; i += 256) ru32[i] = 0u;

  const float gamma = gamma_p[0];
  const float a1 = alphas_p[0], a2 = alphas_p[1], a3 = alphas_p[2];
  const int steps = steps_p[0];
  float whb[8];
#pragma unroll
  for (int nt = 0; nt < 8; ++nt) whb[nt] = Wh_b[nbase + nt * 16 + l16];
  __syncthreads();

  // ---- 30-step recurrence: ring slot (t&3) is s^(t); slots (t+3/2/1)&3
  //      hold s^(t-1..t-3); s^(t+1) overwrites the dropped slot (t+1)&3. --
  for (int t = 0; t < steps; ++t) {
    const unsigned short* sl  = &ring[(t & 3) * SLOT_USHORTS];
    const unsigned short* h1s = &ring[((t + 3) & 3) * SLOT_USHORTS];
    const unsigned short* h2s = &ring[((t + 2) & 3) * SLOT_USHORTS];
    const unsigned short* h3s = &ring[((t + 1) & 3) * SLOT_USHORTS];
    unsigned short*       wr  = &ring[((t + 1) & 3) * SLOT_USHORTS];

#pragma unroll
    for (int nt = 0; nt < 8; ++nt) acc[nt] = v8f{};

    for (int kc = 0; kc < 16; ++kc) {
      v16bf a = load_a_frag(sl + arow * LDS_STRIDE + kc * 32 + lh * 16);
#pragma unroll
      for (int nt = 0; nt < 8; ++nt) {
        const __bf16* bp = (const __bf16*)Wh_bf +
                           (nbase + nt * 16 + l16) * H_DIM + kc * 32 + lh * 16;
        acc[nt] = __builtin_amdgcn_wmma_f32_16x16x32_bf16(
            false, a, false, *(const v16bf*)bp, (short)0, acc[nt], false, false);
      }
    }

    // elementwise: pre = acc + xproj + Wh_b + gamma*recirc; state update.
    // Each (m,n) element of slots is touched by exactly one lane this step;
    // the h3 read precedes the wr write to the same address in-lane (DS in-order).
#pragma unroll
    for (int nt = 0; nt < 8; ++nt) {
#pragma unroll
      for (int r = 0; r < 8; ++r) {
        const int m   = mstrip * 16 + lh * 8 + r;       // C/D layout (ISA 7.12.2)
        const int n   = nbase + nt * 16 + l16;
        const int off = m * LDS_STRIDE + n;
        float rec = a1 * bf2f(h1s[off]) + a2 * bf2f(h2s[off]) + a3 * bf2f(h3s[off]);
        float s   = bf2f(sl[off]);
        float pre = acc[nt][r] + xp[nt][r] + whb[nt] + gamma * rec;
        float hn  = fast_tanh(pre);
        wr[off]   = f2bf(0.5f * s + 0.5f * hn);         // ALPHA = 0.5
      }
    }
    __syncthreads();
  }

  // ---- head: out = s^(steps) @ Head^T + Head_b (N = 256) ---------------
  const unsigned short* sf = &ring[(steps & 3) * SLOT_USHORTS];
  const int nbase2 = (wave >> 1) * 64;      // 2 mstrips x 4 blocks of 64 = 32x256
  v8f acc2[4];
#pragma unroll
  for (int nt = 0; nt < 4; ++nt) acc2[nt] = v8f{};

  for (int kc = 0; kc < 16; ++kc) {
    v16bf a = load_a_frag(sf + arow * LDS_STRIDE + kc * 32 + lh * 16);
#pragma unroll
    for (int nt = 0; nt < 4; ++nt) {
      const __bf16* bp = (const __bf16*)Hd_bf +
                         (nbase2 + nt * 16 + l16) * H_DIM + kc * 32 + lh * 16;
      acc2[nt] = __builtin_amdgcn_wmma_f32_16x16x32_bf16(
          false, a, false, *(const v16bf*)bp, (short)0, acc2[nt], false, false);
    }
  }

#pragma unroll
  for (int nt = 0; nt < 4; ++nt) {
    const int n = nbase2 + nt * 16 + l16;
    float bias = Hd_b[n];
#pragma unroll
    for (int r = 0; r < 8; ++r) {
      const int m = mstrip * 16 + lh * 8 + r;
      out[(size_t)(wg * MTILE + m) * OUT_DIM + n] = acc2[nt][r] + bias;
    }
  }
}

extern "C" void kernel_launch(void* const* d_in, const int* in_sizes, int n_in,
                              void* d_out, int out_size, void* d_ws, size_t ws_size,
                              hipStream_t stream) {
  (void)in_sizes; (void)n_in; (void)out_size;

  const float* x      = (const float*)d_in[0];
  const float* Wx_w   = (const float*)d_in[1];
  const float* Wx_b   = (const float*)d_in[2];
  const float* Wh_w   = (const float*)d_in[3];
  const float* Wh_b   = (const float*)d_in[4];
  const float* Head_w = (const float*)d_in[5];
  const float* Head_b = (const float*)d_in[6];
  const float* gamma  = (const float*)d_in[7];
  const float* alphas = (const float*)d_in[8];
  const int*   steps  = (const int*)d_in[9];

  const int nWx = H_DIM * IN_DIM;           // 262144
  const int nWh = H_DIM * H_DIM;            // 262144
  const int nHd = OUT_DIM * H_DIM;          // 131072
  if (ws_size < (size_t)(nWx + nWh + nHd) * sizeof(unsigned short)) return;

  unsigned short* Wx_bf = (unsigned short*)d_ws;
  unsigned short* Wh_bf = Wx_bf + nWx;
  unsigned short* Hd_bf = Wh_bf + nWh;

  weights_to_bf16<<<(nWx + 255) / 256, 256, 0, stream>>>(Wx_w, Wx_bf, nWx);
  weights_to_bf16<<<(nWh + 255) / 256, 256, 0, stream>>>(Wh_w, Wh_bf, nWh);
  weights_to_bf16<<<(nHd + 255) / 256, 256, 0, stream>>>(Head_w, Hd_bf, nHd);

  toroidal_mlp_kernel<<<8192 / MTILE, 256, RING_BYTES, stream>>>(
      x, Wx_bf, Wx_b, Wh_bf, Wh_b, Hd_bf, Head_b, gamma, alphas, steps,
      (float*)d_out);
}
